// PaPs_11433202942249
// MI455X (gfx1250) — matrix operand order repaired and checked by reading, not compiled, not run on hardware
//
#include <hip/hip_runtime.h>
#include <math.h>

// ---------------------------------------------------------------- constants
#define BATCH   4
#define HH      256
#define WW      256
#define HWSZ    65536          // 256*256
#define NPIX    262144         // 4*256*256
#define ENC     128
#define STACK   896
#define HD      448
#define QD      224
#define KSEL    1024
#define PP      32
#define NCLS    20
#define EPSV    1e-5f

// output layout (floats, concatenated in return order)
#define OUT_HM    0
#define OUT_SAL   262144
#define OUT_SEM   524288       // 1024*20
#define OUT_SIZE  544768       // 1024*2
#define OUT_CONF  546816       // 1024
#define OUT_MASK  547840       // 1024*32*32

// workspace layout (byte offsets)
#define OFF_HHM    0u            // bf16 4*32*256*256
#define OFF_HSAL   16777216u     // bf16
#define OFF_KEYS   33554432u     // u32 262144
#define OFF_STATE  34603008u     // RState
#define OFF_SEL    34605056u     // int[1024]
#define OFF_META   34609152u     // int[3*1024]
#define OFF_STACK  34621440u     // bf16 1024*896
#define OFF_SHH    36456448u     // bf16 1024*448
#define OFF_SZ1    37373952u     // bf16 1024*448
#define OFF_CL1    38291456u     // bf16 1024*448
#define OFF_SZ2    39208960u     // bf16 1024*224
#define OFF_CL2    39667712u     // bf16 1024*224
#define OFF_SHAPES 40126464u     // f32 1024*256
#define OFF_WHM1   41175040u     // bf16 9*32*128  [tap][n][cin]
#define OFF_WSAL1  41248768u     // bf16 9*32*128
#define OFF_WSHP1  41322496u     // bf16 448*896
#define OFF_WSHP2  42125312u     // bf16 256*448
#define OFF_WSZ1   42354688u     // bf16 448*896
#define OFF_WSZ2   43157504u     // bf16 224*448
#define OFF_WSZ3   43358208u     // bf16 2*224 (padded)
#define OFF_WCL1   43359232u     // bf16 448*896
#define OFF_WCL2   44162048u     // bf16 224*448
#define OFF_WCL3   44362752u     // bf16 20*224 (padded)
#define OFF_XBF    44371968u     // bf16 NHWC 4*256*256*128 = 67108864 B

struct RState {
  unsigned prefixBits;
  unsigned maskHigh;
  int      needK;
  int      cntH;
  int      cntE;
  float    meanAcc;
  unsigned hist[256];
};

// ---------------------------------------------------------------- helpers
typedef __attribute__((ext_vector_type(16))) __bf16 v16bf;
typedef __attribute__((ext_vector_type(8)))  __bf16 v8bf;
typedef __attribute__((ext_vector_type(8)))  float  v8f;

__device__ __forceinline__ v16bf load16(const __bf16* p) {
  v8bf lo = *(const v8bf*)p;          // 16B
  v8bf hi = *(const v8bf*)(p + 16);   // 16B
  return __builtin_shufflevector(lo, hi, 0,1,2,3,4,5,6,7,8,9,10,11,12,13,14,15);
}
__device__ __forceinline__ int refl(int i) { return i < 0 ? -i : (i > 255 ? 510 - i : i); }
__device__ __forceinline__ float sigmoidf(float x) { return 1.0f / (1.0f + expf(-x)); }
__device__ __forceinline__ unsigned sortkey(float f) {
  unsigned b = __float_as_uint(f);
  return (b & 0x80000000u) ? ~b : (b | 0x80000000u);
}

// ================================================================ init
__global__ __launch_bounds__(256) void init_state(RState* st) {
  int t = threadIdx.x;
  if (t == 0) {
    st->prefixBits = 0; st->maskHigh = 0; st->needK = KSEL;
    st->cntH = 0; st->cntE = 0; st->meanAcc = 0.0f;
  }
  st->hist[t] = 0u;
}

// ================================================================ pre-convert kernels
// x (NCHW f32) -> xbf (NHWC bf16); reads coalesced per channel iteration
__global__ __launch_bounds__(256) void cvt_x_nhwc(const float* __restrict__ x,
                                                  __bf16* __restrict__ xbf)
{
  const int pix = blockIdx.x * 256 + threadIdx.x;   // 0..NPIX-1
  const int b = pix >> 16, hw = pix & 65535;
  __bf16* dp = xbf + (size_t)pix * ENC;
  const float* sp = x + ((size_t)b * ENC << 16) + hw;
#pragma unroll 4
  for (int c = 0; c < ENC; ++c) dp[c] = (__bf16)sp[(size_t)c << 16];
}

// head conv weight (32,128,3,3) OIHW -> bf16 [tap][n][cin]
__global__ __launch_bounds__(256) void cvt_w_head(const float* __restrict__ w,
                                                  __bf16* __restrict__ wb)
{
  const int i = blockIdx.x * 256 + threadIdx.x;     // 0..36863
  const int tap = i >> 12, r = i & 4095, n = r >> 7, c = r & 127;
  wb[i] = (__bf16)w[n * 1152 + c * 9 + tap];
}

// generic f32 -> bf16 copy
__global__ __launch_bounds__(256) void cvt_f32_bf16(const float* __restrict__ s,
                                                    __bf16* __restrict__ d, int n)
{
  const int i = blockIdx.x * 256 + threadIdx.x;
  if (i < n) d[i] = (__bf16)s[i];
}

// ================================================================ conv1 (128->32, reflect pad, BN+ReLU) via WMMA bf16
// grid = 2048 x 256thr (8 waves); wave = 16 pixels (M) x 32 outch (2 N tiles)
__global__ __launch_bounds__(256) void head_conv1_wmma(
    const __bf16* __restrict__ xbf,    // NHWC bf16
    const __bf16* __restrict__ wb,     // [tap][n][cin] bf16
    const float* __restrict__ bias, const float* __restrict__ g,
    const float* __restrict__ be, const float* __restrict__ mu,
    const float* __restrict__ var, __bf16* __restrict__ out)
{
  const int lane = threadIdx.x & 31;
  const int wave = threadIdx.x >> 5;
  const int half = lane >> 4;
  const int mrow = lane & 15;
  const int nB   = lane & 15;
  const int mbase = (blockIdx.x * 8 + wave) * 16;

  const int m    = mbase + mrow;
  const int bimg = m >> 16;
  const int h0   = (m >> 8) & 255;
  const int w0   = m & 255;

  v8f acc0 = {}; v8f acc1 = {};

  for (int tap = 0; tap < 9; ++tap) {
    const int hh = refl(h0 + tap / 3 - 1), xx = refl(w0 + tap % 3 - 1);
    const __bf16* xp = xbf + (size_t)((bimg << 16) + hh * WW + xx) * ENC;
    const __bf16* wp0 = wb + (size_t)(tap * 32 + nB) * ENC + half * 8;
    const __bf16* wp1 = wp0 + (size_t)16 * ENC;
#pragma unroll
    for (int c0 = 0; c0 < 128; c0 += 32) {
      // A: two contiguous 8-elem runs -> 2x b128 loads
      v8bf alo = *(const v8bf*)(xp + c0 + half * 8);
      v8bf ahi = *(const v8bf*)(xp + c0 + 16 + half * 8);
      v16bf av = __builtin_shufflevector(alo, ahi, 0,1,2,3,4,5,6,7,8,9,10,11,12,13,14,15);
      v16bf b0 = load16(wp0 + c0);
      v16bf b1 = load16(wp1 + c0);
      acc0 = __builtin_amdgcn_wmma_f32_16x16x32_bf16(false, av, false, b0, (short)0, acc0, false, false);
      acc1 = __builtin_amdgcn_wmma_f32_16x16x32_bf16(false, av, false, b1, (short)0, acc1, false, false);
    }
  }

#pragma unroll
  for (int t = 0; t < 2; ++t) {
    const int n = nB + t * 16;
    const float inv  = g[n] / sqrtf(var[n] + EPSV);
    const float beta = bias[n] * inv + be[n] - mu[n] * inv;
    v8f acc = t ? acc1 : acc0;
#pragma unroll
    for (int v = 0; v < 8; ++v) {
      const int mm = mbase + v + half * 8;
      const int bb = mm >> 16;
      out[((size_t)(bb * 32 + n) << 16) + (mm & 65535)] =
          (__bf16)fmaxf(acc[v] * inv + beta, 0.0f);
    }
  }
}

// ================================================================ conv2 (32->1, reflect pad, BN, optional sigmoid)
__global__ __launch_bounds__(256) void head_conv2(
    const __bf16* __restrict__ hin, const float* __restrict__ w, // (1,32,3,3)
    const float* __restrict__ bias, const float* __restrict__ g,
    const float* __restrict__ be, const float* __restrict__ mu,
    const float* __restrict__ var, float* __restrict__ out, int do_sig)
{
  const int i = blockIdx.x * 256 + threadIdx.x;
  const int b = i >> 16, h = (i >> 8) & 255, ww = i & 255;
  float s = 0.0f;
  for (int c = 0; c < 32; ++c) {
    const __bf16* hp = hin + ((size_t)(b * 32 + c) << 16);
#pragma unroll
    for (int tap = 0; tap < 9; ++tap) {
      const int hh = refl(h + tap / 3 - 1), xx = refl(ww + tap % 3 - 1);
      s += (float)hp[hh * WW + xx] * w[c * 9 + tap];
    }
  }
  s += bias[0];
  const float inv = g[0] / sqrtf(var[0] + EPSV);
  s = s * inv + be[0] - mu[0] * inv;
  if (do_sig) s = sigmoidf(s);
  out[i] = s;
}

// ================================================================ global mean of heatmap
__global__ __launch_bounds__(256) void reduce_mean(const float* __restrict__ hm, RState* st) {
  __shared__ float red[256];
  const int t = threadIdx.x;
  const int i0 = blockIdx.x * 1024 + t;
  float s = 0.0f;
#pragma unroll
  for (int j = 0; j < 4; ++j) s += hm[i0 + 256 * j];
  red[t] = s; __syncthreads();
  for (int k = 128; k > 0; k >>= 1) { if (t < k) red[t] += red[t + k]; __syncthreads(); }
  if (t == 0) atomicAdd(&st->meanAcc, red[0]);
}

// ================================================================ NMS scores -> sortable keys
__global__ __launch_bounds__(256) void scores_kernel(const float* __restrict__ hm,
                                                     const RState* st,
                                                     unsigned* __restrict__ keys)
{
  const float mean = st->meanAcc * (1.0f / (float)NPIX);
  const int i0 = blockIdx.x * 1024 + threadIdx.x;
#pragma unroll
  for (int j = 0; j < 4; ++j) {
    const int i = i0 + 256 * j;
    const int b = i >> 16, h = (i >> 8) & 255, w = i & 255;
    const float v = hm[i];
    float mx = -INFINITY;
    for (int dy = -1; dy <= 1; ++dy) {
      const int hh = h + dy; if (hh < 0 || hh > 255) continue;
      for (int dx = -1; dx <= 1; ++dx) {
        const int ww = w + dx; if (ww < 0 || ww > 255) continue;
        mx = fmaxf(mx, hm[(b << 16) + hh * WW + ww]);
      }
    }
    const float sc = (v == mx && v > mean) ? v : -INFINITY;
    keys[i] = sortkey(sc);
  }
}

// ================================================================ exact top-K radix select (4 x 8-bit MSB passes)
__global__ __launch_bounds__(256) void radix_hist(const unsigned* __restrict__ keys,
                                                  RState* st, int shift)
{
  __shared__ unsigned lh[256];
  const int t = threadIdx.x;
  lh[t] = 0u; __syncthreads();
  const unsigned pb = st->prefixBits, mk = st->maskHigh;
  const int i0 = blockIdx.x * 1024 + t;
#pragma unroll
  for (int j = 0; j < 4; ++j) {
    const unsigned k = keys[i0 + 256 * j];
    if ((k & mk) == pb) atomicAdd(&lh[(k >> shift) & 255u], 1u);
  }
  __syncthreads();
  atomicAdd(&st->hist[t], lh[t]);
}

__global__ __launch_bounds__(256) void radix_resolve(RState* st, int shift) {
  const int t = threadIdx.x;
  if (t == 0) {
    const int need = st->needK;
    unsigned cum = 0; int bsel = 0;
    for (int b = 255; b >= 0; --b) {
      const unsigned c = st->hist[b];
      if (cum + c >= (unsigned)need) { bsel = b; st->needK = need - (int)cum; break; }
      cum += c;
    }
    st->prefixBits |= ((unsigned)bsel) << shift;
    st->maskHigh   |= 0xffu << shift;
  }
  __syncthreads();
  st->hist[t] = 0u;
}

__global__ __launch_bounds__(256) void compact_topk(const unsigned* __restrict__ keys,
                                                    RState* st, int* __restrict__ sel)
{
  const unsigned T = st->prefixBits;
  const int need = st->needK;
  const int i0 = blockIdx.x * 1024 + threadIdx.x;
#pragma unroll
  for (int j = 0; j < 4; ++j) {
    const int i = i0 + 256 * j;
    const unsigned k = keys[i];
    if (k > T) {
      sel[atomicAdd(&st->cntH, 1)] = i;
    } else if (k == T) {
      const int p = atomicAdd(&st->cntE, 1);
      if (p < need) sel[KSEL - need + p] = i;
    }
  }
}

// ================================================================ gather features (bf16 stack) / confidence / meta
__global__ __launch_bounds__(256) void gather_kernel(
    const int* __restrict__ sel,
    const float* __restrict__ f0, const float* __restrict__ f1, const float* __restrict__ f2,
    const float* __restrict__ hm,
    __bf16* __restrict__ stack, int* __restrict__ meta, float* __restrict__ conf)
{
  const int k = blockIdx.x;
  const int idx = sel[k];
  const int cb = idx >> 16, ch = (idx >> 8) & 255, cw = idx & 255;
  if (threadIdx.x == 0) {
    meta[k * 3 + 0] = cb; meta[k * 3 + 1] = ch; meta[k * 3 + 2] = cw;
    conf[k] = hm[idx];
  }
  for (int c = threadIdx.x; c < STACK; c += 256) {
    float v;
    if (c < 128)       v = f0[((size_t)(cb * 128 + c) << 16) + (ch << 8) + cw];
    else if (c < 384)  v = f1[(size_t)(cb * 256 + (c - 128)) * 16384 + (ch >> 1) * 128 + (cw >> 1)];
    else               v = f2[(size_t)(cb * 512 + (c - 384)) * 4096 + (ch >> 2) * 64 + (cw >> 2)];
    stack[(size_t)k * STACK + c] = (__bf16)v;
  }
}

// ================================================================ WMMA GEMM: C = act(A @ W^T + bias)
// A: M x K bf16 row-major; Wt: N x K bf16 row-major
// act: 0 bias, 1 bias+BN+ReLU, 2 bias+softplus ; out_bf16 selects output type
__global__ __launch_bounds__(256) void gemm_wmma(
    const __bf16* __restrict__ A, const __bf16* __restrict__ Wt,
    const float* __restrict__ bias, const float* __restrict__ gb,
    const float* __restrict__ beb, const float* __restrict__ mub,
    const float* __restrict__ vb,
    void* __restrict__ Cout, int Kd, int N, int act, int out_bf16)
{
  const int lane = threadIdx.x & 31;
  const int wave = threadIdx.x >> 5;
  const int half = lane >> 4;
  const int mrow = lane & 15;
  const int tm = (blockIdx.x * 8 + wave) * 16;
  const int tn = blockIdx.y * 16;
  const int n  = tn + (lane & 15);
  const bool nok = (n < N);

  v8f acc = {};
  const __bf16* ap = A + (size_t)(tm + mrow) * Kd;
  const __bf16* wp = Wt + (size_t)(nok ? n : 0) * Kd + half * 8;
  for (int k0 = 0; k0 < Kd; k0 += 32) {
    v8bf alo = *(const v8bf*)(ap + k0 + half * 8);
    v8bf ahi = *(const v8bf*)(ap + k0 + 16 + half * 8);
    v16bf av = __builtin_shufflevector(alo, ahi, 0,1,2,3,4,5,6,7,8,9,10,11,12,13,14,15);
    v16bf bv = load16(wp + k0);
    acc = __builtin_amdgcn_wmma_f32_16x16x32_bf16(false, av, false, bv, (short)0, acc, false, false);
  }

  if (!nok) return;
  float inv = 1.0f, beta = bias[n];
  if (act == 1) {
    inv  = gb[n] / sqrtf(vb[n] + EPSV);
    beta = bias[n] * inv + beb[n] - mub[n] * inv;
  }
#pragma unroll
  for (int v = 0; v < 8; ++v) {
    const int mm = tm + v + half * 8;
    float val = acc[v] * inv + beta;
    if (act == 1) val = fmaxf(val, 0.0f);
    else if (act == 2) val = (val > 20.0f) ? val : log1pf(expf(val));
    if (out_bf16) ((__bf16*)Cout)[(size_t)mm * N + n] = (__bf16)val;
    else          ((float*)Cout)[(size_t)mm * N + n]  = val;
  }
}

// ================================================================ per-peak mask head
__global__ __launch_bounds__(256) void mask_kernel(
    const float* __restrict__ shapes, const float* __restrict__ sal,
    const int* __restrict__ meta,
    const float* __restrict__ w1, const float* __restrict__ b1,
    const float* __restrict__ g,  const float* __restrict__ be,
    const float* __restrict__ w2, const float* __restrict__ b2,
    const float* __restrict__ w3, const float* __restrict__ b3,
    float* __restrict__ masks)
{
  __shared__ float  ssh[256];
  __shared__ float  sm[1024];
  __shared__ __bf16 sh1[16 * 1024];
  __shared__ float  plane[1024];
  __shared__ float  oacc[1024];
  __shared__ float  red[256];

  const int k = blockIdx.x, tid = threadIdx.x;
  const int cb = meta[k * 3 + 0], ch = meta[k * 3 + 1], cw = meta[k * 3 + 2];
  ssh[tid] = shapes[(size_t)k * 256 + tid];
  __syncthreads();

  const int hs  = min(max(ch - 16, 0), HH - PP);
  const int wsv = min(max(cw - 16, 0), WW - PP);

  for (int p = tid; p < 1024; p += 256) {
    const int y = p >> 5, x = p & 31;
    const float fy = (y + 0.5f) * 0.5f - 0.5f;
    const float fx = (x + 0.5f) * 0.5f - 0.5f;
    const int y0 = (int)floorf(fy), x0 = (int)floorf(fx);
    const float wy = fy - y0, wx = fx - x0;
    const int y0c = min(max(y0, 0), 15), y1c = min(max(y0 + 1, 0), 15);
    const int x0c = min(max(x0, 0), 15), x1c = min(max(x0 + 1, 0), 15);
    const float up = (1 - wy) * ((1 - wx) * ssh[y0c * 16 + x0c] + wx * ssh[y0c * 16 + x1c])
                   +      wy  * ((1 - wx) * ssh[y1c * 16 + x0c] + wx * ssh[y1c * 16 + x1c]);
    const float cr = sal[(size_t)cb * HWSZ + (hs + y) * WW + (wsv + x)];
    sm[p] = up * cr;
  }
  __syncthreads();

  float lsum = 0.0f, lsq = 0.0f;
  for (int t = tid; t < 16384; t += 256) {
    const int c = t >> 10, p = t & 1023, y = p >> 5, x = p & 31;
    float s = b1[c];
#pragma unroll
    for (int tap = 0; tap < 9; ++tap) {
      const int yy = y + tap / 3 - 1, xx = x + tap % 3 - 1;
      if (yy >= 0 && yy < 32 && xx >= 0 && xx < 32) s += w1[c * 9 + tap] * sm[yy * 32 + xx];
    }
    sh1[t] = (__bf16)s; lsum += s; lsq += s * s;
  }
  red[tid] = lsum; __syncthreads();
  for (int s2 = 128; s2 > 0; s2 >>= 1) { if (tid < s2) red[tid] += red[tid + s2]; __syncthreads(); }
  const float mean = red[0] * (1.0f / 16384.0f);
  __syncthreads();
  red[tid] = lsq; __syncthreads();
  for (int s2 = 128; s2 > 0; s2 >>= 1) { if (tid < s2) red[tid] += red[tid + s2]; __syncthreads(); }
  const float varv = red[0] * (1.0f / 16384.0f) - mean * mean;
  const float rinv = rsqrtf(varv + EPSV);
  __syncthreads();

  for (int t = tid; t < 16384; t += 256) {
    const int c = t >> 10;
    float hv = ((float)sh1[t] - mean) * rinv * g[c] + be[c];
    sh1[t] = (__bf16)fmaxf(hv, 0.0f);
  }
  for (int p = tid; p < 1024; p += 256) oacc[p] = b3[0];
  __syncthreads();

  for (int c2 = 0; c2 < 16; ++c2) {
    for (int p = tid; p < 1024; p += 256) {
      const int y = p >> 5, x = p & 31;
      float s = b2[c2];
      for (int c1 = 0; c1 < 16; ++c1) {
        const __bf16* hp = &sh1[c1 * 1024];
        const float* wp = &w2[(c2 * 16 + c1) * 9];
#pragma unroll
        for (int tap = 0; tap < 9; ++tap) {
          const int yy = y + tap / 3 - 1, xx = x + tap % 3 - 1;
          if (yy >= 0 && yy < 32 && xx >= 0 && xx < 32) s += wp[tap] * (float)hp[yy * 32 + xx];
        }
      }
      plane[p] = fmaxf(s, 0.0f);
    }
    __syncthreads();
    for (int p = tid; p < 1024; p += 256) {
      const int y = p >> 5, x = p & 31;
      float s = 0.0f;
#pragma unroll
      for (int tap = 0; tap < 9; ++tap) {
        const int yy = y + tap / 3 - 1, xx = x + tap % 3 - 1;
        if (yy >= 0 && yy < 32 && xx >= 0 && xx < 32) s += w3[c2 * 9 + tap] * plane[yy * 32 + xx];
      }
      oacc[p] += s;
    }
    __syncthreads();
  }

  for (int p = tid; p < 1024; p += 256)
    masks[(size_t)k * 1024 + p] = sigmoidf(sm[p] + oacc[p]);
}

// ================================================================ input index map (top level insertion order; params pytree alphabetical)
enum {
  IN_X = 0, IN_F0, IN_F1, IN_F2,
  CLS_B1 = 4, CLS_B2, CLS_B3, CLS_BE1, CLS_G1, CLS_M1, CLS_V1, CLS_W1, CLS_W2, CLS_W3,
  HM_B1 = 14, HM_B2, HM_BE1, HM_BE2, HM_G1, HM_G2, HM_M1, HM_M2, HM_V1, HM_V2, HM_W1, HM_W2,
  MK_B1 = 26, MK_B2, MK_B3, MK_BE, MK_G, MK_W1, MK_W2, MK_W3,
  SAL_B1 = 34, SAL_B2, SAL_BE1, SAL_BE2, SAL_G1, SAL_G2, SAL_M1, SAL_M2, SAL_V1, SAL_V2, SAL_W1, SAL_W2,
  SHP_B1 = 46, SHP_B2, SHP_BE1, SHP_G1, SHP_M1, SHP_V1, SHP_W1, SHP_W2,
  SZ_B1 = 54, SZ_B2, SZ_B3, SZ_BE1, SZ_BE2, SZ_G1, SZ_G2, SZ_M1, SZ_M2, SZ_V1, SZ_V2, SZ_W1, SZ_W2, SZ_W3
};

extern "C" void kernel_launch(void* const* d_in, const int* in_sizes, int n_in,
                              void* d_out, int out_size, void* d_ws, size_t ws_size,
                              hipStream_t stream) {
  (void)in_sizes; (void)n_in; (void)out_size; (void)ws_size;
  #define FIN(i) ((const float*)d_in[(i)])
  float* out = (float*)d_out;
  char* ws = (char*)d_ws;

  __bf16*   hhm  = (__bf16*)(ws + OFF_HHM);
  __bf16*   hsal = (__bf16*)(ws + OFF_HSAL);
  unsigned* keys = (unsigned*)(ws + OFF_KEYS);
  RState*   st   = (RState*)(ws + OFF_STATE);
  int*      sel  = (int*)(ws + OFF_SEL);
  int*      meta = (int*)(ws + OFF_META);
  __bf16*   stk  = (__bf16*)(ws + OFF_STACK);
  __bf16*   shh  = (__bf16*)(ws + OFF_SHH);
  __bf16*   sz1  = (__bf16*)(ws + OFF_SZ1);
  __bf16*   cl1  = (__bf16*)(ws + OFF_CL1);
  __bf16*   sz2  = (__bf16*)(ws + OFF_SZ2);
  __bf16*   cl2  = (__bf16*)(ws + OFF_CL2);
  float*    shp  = (float*)(ws + OFF_SHAPES);
  __bf16*   whm1 = (__bf16*)(ws + OFF_WHM1);
  __bf16*   wsl1 = (__bf16*)(ws + OFF_WSAL1);
  __bf16*   wsp1 = (__bf16*)(ws + OFF_WSHP1);
  __bf16*   wsp2 = (__bf16*)(ws + OFF_WSHP2);
  __bf16*   wsz1 = (__bf16*)(ws + OFF_WSZ1);
  __bf16*   wsz2 = (__bf16*)(ws + OFF_WSZ2);
  __bf16*   wsz3 = (__bf16*)(ws + OFF_WSZ3);
  __bf16*   wcl1 = (__bf16*)(ws + OFF_WCL1);
  __bf16*   wcl2 = (__bf16*)(ws + OFF_WCL2);
  __bf16*   wcl3 = (__bf16*)(ws + OFF_WCL3);
  __bf16*   xbf  = (__bf16*)(ws + OFF_XBF);

  init_state<<<1, 256, 0, stream>>>(st);

  // --- operand pre-conversion to bf16 (WMMA-friendly layouts) ---
  cvt_x_nhwc<<<NPIX / 256, 256, 0, stream>>>(FIN(IN_X), xbf);
  cvt_w_head<<<144, 256, 0, stream>>>(FIN(HM_W1), whm1);
  cvt_w_head<<<144, 256, 0, stream>>>(FIN(SAL_W1), wsl1);
  cvt_f32_bf16<<<(HD * STACK + 255) / 256, 256, 0, stream>>>(FIN(SHP_W1), wsp1, HD * STACK);
  cvt_f32_bf16<<<(256 * HD + 255) / 256, 256, 0, stream>>>(FIN(SHP_W2), wsp2, 256 * HD);
  cvt_f32_bf16<<<(HD * STACK + 255) / 256, 256, 0, stream>>>(FIN(SZ_W1), wsz1, HD * STACK);
  cvt_f32_bf16<<<(QD * HD + 255) / 256, 256, 0, stream>>>(FIN(SZ_W2), wsz2, QD * HD);
  cvt_f32_bf16<<<(2 * QD + 255) / 256, 256, 0, stream>>>(FIN(SZ_W3), wsz3, 2 * QD);
  cvt_f32_bf16<<<(HD * STACK + 255) / 256, 256, 0, stream>>>(FIN(CLS_W1), wcl1, HD * STACK);
  cvt_f32_bf16<<<(QD * HD + 255) / 256, 256, 0, stream>>>(FIN(CLS_W2), wcl2, QD * HD);
  cvt_f32_bf16<<<(NCLS * QD + 255) / 256, 256, 0, stream>>>(FIN(CLS_W3), wcl3, NCLS * QD);

  // --- heads: conv1 (WMMA) + conv2 ---
  head_conv1_wmma<<<2048, 256, 0, stream>>>(xbf, whm1, FIN(HM_B1),
      FIN(HM_G1), FIN(HM_BE1), FIN(HM_M1), FIN(HM_V1), hhm);
  head_conv1_wmma<<<2048, 256, 0, stream>>>(xbf, wsl1, FIN(SAL_B1),
      FIN(SAL_G1), FIN(SAL_BE1), FIN(SAL_M1), FIN(SAL_V1), hsal);
  head_conv2<<<1024, 256, 0, stream>>>(hhm, FIN(HM_W2), FIN(HM_B2),
      FIN(HM_G2), FIN(HM_BE2), FIN(HM_M2), FIN(HM_V2), out + OUT_HM, 1);
  head_conv2<<<1024, 256, 0, stream>>>(hsal, FIN(SAL_W2), FIN(SAL_B2),
      FIN(SAL_G2), FIN(SAL_BE2), FIN(SAL_M2), FIN(SAL_V2), out + OUT_SAL, 0);

  // --- peak scores + exact top-1024 radix select ---
  reduce_mean<<<256, 256, 0, stream>>>(out + OUT_HM, st);
  scores_kernel<<<256, 256, 0, stream>>>(out + OUT_HM, st, keys);
  for (int p = 0; p < 4; ++p) {
    const int shift = 24 - 8 * p;
    radix_hist<<<256, 256, 0, stream>>>(keys, st, shift);
    radix_resolve<<<1, 256, 0, stream>>>(st, shift);
  }
  compact_topk<<<256, 256, 0, stream>>>(keys, st, sel);

  gather_kernel<<<KSEL, 256, 0, stream>>>(sel, FIN(IN_F0), FIN(IN_F1), FIN(IN_F2),
      out + OUT_HM, stk, meta, out + OUT_CONF);

  // --- shape MLP: 896 -> 448 (BN+ReLU) -> 256 ---
  gemm_wmma<<<dim3(8, 28), 256, 0, stream>>>(stk, wsp1, FIN(SHP_B1),
      FIN(SHP_G1), FIN(SHP_BE1), FIN(SHP_M1), FIN(SHP_V1), shh, STACK, HD, 1, 1);
  gemm_wmma<<<dim3(8, 16), 256, 0, stream>>>(shh, wsp2, FIN(SHP_B2),
      nullptr, nullptr, nullptr, nullptr, shp, HD, 256, 0, 0);

  // --- size MLP: 896 -> 448 (BN+ReLU) -> 224 (BN+ReLU) -> 2 (softplus) ---
  gemm_wmma<<<dim3(8, 28), 256, 0, stream>>>(stk, wsz1, FIN(SZ_B1),
      FIN(SZ_G1), FIN(SZ_BE1), FIN(SZ_M1), FIN(SZ_V1), sz1, STACK, HD, 1, 1);
  gemm_wmma<<<dim3(8, 14), 256, 0, stream>>>(sz1, wsz2, FIN(SZ_B2),
      FIN(SZ_G2), FIN(SZ_BE2), FIN(SZ_M2), FIN(SZ_V2), sz2, HD, QD, 1, 1);
  gemm_wmma<<<dim3(8, 1), 256, 0, stream>>>(sz2, wsz3, FIN(SZ_B3),
      nullptr, nullptr, nullptr, nullptr, out + OUT_SIZE, QD, 2, 2, 0);

  // --- class MLP: 896 -> 448 (BN+ReLU) -> 224 -> 20 ---
  gemm_wmma<<<dim3(8, 28), 256, 0, stream>>>(stk, wcl1, FIN(CLS_B1),
      FIN(CLS_G1), FIN(CLS_BE1), FIN(CLS_M1), FIN(CLS_V1), cl1, STACK, HD, 1, 1);
  gemm_wmma<<<dim3(8, 14), 256, 0, stream>>>(cl1, wcl2, FIN(CLS_B2),
      nullptr, nullptr, nullptr, nullptr, cl2, HD, QD, 0, 1);
  gemm_wmma<<<dim3(8, 2), 256, 0, stream>>>(cl2, wcl3, FIN(CLS_B3),
      nullptr, nullptr, nullptr, nullptr, out + OUT_SEM, QD, NCLS, 0, 0);

  // --- per-peak masks ---
  mask_kernel<<<KSEL, 256, 0, stream>>>(shp, out + OUT_SAL, meta,
      FIN(MK_W1), FIN(MK_B1), FIN(MK_G), FIN(MK_BE),
      FIN(MK_W2), FIN(MK_B2), FIN(MK_W3), FIN(MK_B3), out + OUT_MASK);
  #undef FIN
}
